// CVTransformer_21277267985090
// MI455X (gfx1250) — matrix operand order repaired and checked by reading, not compiled, not run on hardware
//
#include <hip/hip_runtime.h>

#define LAYERS 4
#define DMODEL 256
#define NHEAD 8
#define DHEAD 32
#define WSZ 128
#define FFDIM 1024
#define NBATCH 2
#define SEQ 8192
#define MROWS (NBATCH * SEQ)   /* 16384 */
#define ATT_SCALE 0.17677669529663687f
#define LPAD 40                /* LDS row stride in ushorts: 80B, 16B-aligned, bank-conflict-free */

typedef __attribute__((ext_vector_type(16))) __bf16 v16bf;
typedef __attribute__((ext_vector_type(8)))  float  v8f;

__device__ __forceinline__ unsigned short f2bf(float f) {
  unsigned int u = __float_as_uint(f);
  u += 0x7FFFu + ((u >> 16) & 1u);               // round-to-nearest-even
  return (unsigned short)(u >> 16);
}
__device__ __forceinline__ float bf2f(unsigned short h) {
  return __uint_as_float(((unsigned int)h) << 16);
}

union Frag16 { unsigned int u[8]; v16bf v; };

// ---------------------------------------------------------------------------
// Pack [x_real | x_imag] -> X fp32 [MROWS, 512]
// ---------------------------------------------------------------------------
__global__ void pack_x_kernel(const float* __restrict__ xr,
                              const float* __restrict__ xi,
                              float* __restrict__ X) {
  int idx = blockIdx.x * blockDim.x + threadIdx.x;
  if (idx >= MROWS * DMODEL) return;
  int row = idx / DMODEL, c = idx % DMODEL;
  X[(size_t)row * 512 + c]       = xr[idx];
  X[(size_t)row * 512 + 256 + c] = xi[idx];
}

__global__ void unpack_kernel(const float* __restrict__ X,
                              float* __restrict__ out) {
  int idx = blockIdx.x * blockDim.x + threadIdx.x;
  if (idx >= MROWS * DMODEL) return;
  int row = idx / DMODEL, c = idx % DMODEL;
  out[idx]                  = X[(size_t)row * 512 + c];
  out[MROWS * DMODEL + idx] = X[(size_t)row * 512 + 256 + c];
}

// ---------------------------------------------------------------------------
// Complex weight packing, stored TRANSPOSED as Bt[2*Out (n), 2*In (k)] bf16
// so the GEMM B-tile load is a contiguous K-row copy (same as the A tile).
// Logical packed matrix B[2*In, 2*Out]:
//   cols [0,Out):   rows[0,In)=Wr , rows[In,2In)=-Wi
//   cols [Out,2Out): rows[0,In)=Wi , rows[In,2In)= Wr
// Here we write dst[(colOff + c) * Kdim + r] = B[r][c].
// ---------------------------------------------------------------------------
__global__ void pack2_kernel(const float* __restrict__ Wr,
                             const float* __restrict__ Wi,
                             unsigned short* __restrict__ dst,
                             int In, int Out, int Kdim, int colOff) {
  int idx = blockIdx.x * blockDim.x + threadIdx.x;
  int total = 4 * In * Out;
  if (idx >= total) return;
  int r = idx % (2 * In);          // k index (contiguous in dst row)
  int c = idx / (2 * In);          // n index
  int rin = (r < In) ? r : r - In;
  int j = (c < Out) ? c : c - Out;
  float v;
  if (c < Out) v = (r < In) ? Wr[(size_t)rin * Out + j] : -Wi[(size_t)rin * Out + j];
  else         v = (r < In) ? Wi[(size_t)rin * Out + j] :  Wr[(size_t)rin * Out + j];
  dst[(size_t)(colOff + c) * Kdim + r] = f2bf(v);
}

__global__ void packbias_kernel(const float* __restrict__ br,
                                const float* __restrict__ bi,
                                float* __restrict__ dst, int Out) {
  int i = blockIdx.x * blockDim.x + threadIdx.x;
  if (i >= 2 * Out) return;
  dst[i] = (i < Out) ? (br[i] - bi[i]) : (br[i - Out] + bi[i - Out]);
}

// ---------------------------------------------------------------------------
// Complex RMSNorm: X fp32 [MROWS,512] -> Abuf bf16 [MROWS,512]
// rms = sqrt(sum_{512}(X^2)/256 + eps); scale index = col % 256
// ---------------------------------------------------------------------------
__global__ __launch_bounds__(128) void rmsnorm_kernel(
    const float* __restrict__ X, const float* __restrict__ scale,
    unsigned short* __restrict__ out) {
  __shared__ float red[128];
  int row = blockIdx.x, tid = threadIdx.x;
  const float* xr = X + (size_t)row * 512;
  float x0 = xr[tid], x1 = xr[tid + 128], x2 = xr[tid + 256], x3 = xr[tid + 384];
  red[tid] = x0 * x0 + x1 * x1 + x2 * x2 + x3 * x3;
  __syncthreads();
  for (int s = 64; s > 0; s >>= 1) {
    if (tid < s) red[tid] += red[tid + s];
    __syncthreads();
  }
  float inv = 1.0f / sqrtf(red[0] * (1.0f / 256.0f) + 1e-8f);
  unsigned short* o = out + (size_t)row * 512;
  float s0 = scale[tid], s1 = scale[tid + 128];
  o[tid]       = f2bf(x0 * inv * s0);
  o[tid + 128] = f2bf(x1 * inv * s1);
  o[tid + 256] = f2bf(x2 * inv * s0);
  o[tid + 384] = f2bf(x3 * inv * s1);
}

// ---------------------------------------------------------------------------
// WMMA GEMM: C[M,N] = A[M,K] @ B[K,N], with B supplied TRANSPOSED as Bt[N,K].
// A,Bt bf16; fp32 accum. Block tile 128x128, BK=32, 256 threads = 8 waves
// (4 along M x 2 along N); each wave: 2x4 v_wmma_f32_16x16x32_bf16 per K-step.
// Double-buffered LDS (one barrier per K-step); next tile's global loads are
// issued before the current tile's WMMAs so HBM/L2 latency hides behind math.
// MODE 0: C -> bf16        MODE 1: C+bias, exact GELU -> bf16
// MODE 2: X += C (fp32)    MODE 3: X += C + bias (fp32)   (N==512 for 2/3)
// ---------------------------------------------------------------------------
template <int MODE>
__global__ __launch_bounds__(256) void gemm_kernel(
    const unsigned short* __restrict__ A, const unsigned short* __restrict__ Bt,
    unsigned short* __restrict__ Cbf, float* __restrict__ Xres,
    const float* __restrict__ bias, int N, int K) {
  __shared__ unsigned short As[2][128 * LPAD];
  __shared__ unsigned short Bs[2][128 * LPAD];
  int tid = threadIdx.x;
  int lane = tid & 31;
  int wid = tid >> 5;
  int wm = wid & 3;        // 0..3 -> 32-row slab
  int wn = wid >> 2;       // 0..1 -> 64-col slab
  int bm = blockIdx.y * 128;
  int bn = blockIdx.x * 128;
  int half = lane >> 4;    // lane half selects K sub-range
  int l15 = lane & 15;

  // tile copy role: thread -> row tid>>1 (m for A, n for B), 16-elem half
  int rr = tid >> 1, cc = (tid & 1) * 16;
  const unsigned short* gA = A  + (size_t)(bm + rr) * K + cc;
  const unsigned short* gB = Bt + (size_t)(bn + rr) * K + cc;
  unsigned int ldso = rr * LPAD + cc;

  v8f acc[2][4];
#pragma unroll
  for (int mi = 0; mi < 2; ++mi)
#pragma unroll
    for (int ni = 0; ni < 4; ++ni) {
      v8f z = {0.f, 0.f, 0.f, 0.f, 0.f, 0.f, 0.f, 0.f};
      acc[mi][ni] = z;
    }

  // preload tile 0
  {
    uint4 a0 = *(const uint4*)gA, a1 = *(const uint4*)(gA + 8);
    uint4 b0 = *(const uint4*)gB, b1 = *(const uint4*)(gB + 8);
    *(uint4*)&As[0][ldso] = a0; *(uint4*)&As[0][ldso + 8] = a1;
    *(uint4*)&Bs[0][ldso] = b0; *(uint4*)&Bs[0][ldso + 8] = b1;
  }
  __syncthreads();

  int nk = K >> 5;
  for (int kt = 0; kt < nk; ++kt) {
    int cur = kt & 1, nxt = cur ^ 1;
    bool pre = (kt + 1) < nk;
    uint4 na0, na1, nb0, nb1;
    if (pre) {  // issue next tile's global loads early
      const unsigned short* pa = gA + (kt + 1) * 32;
      const unsigned short* pb = gB + (kt + 1) * 32;
      na0 = *(const uint4*)pa; na1 = *(const uint4*)(pa + 8);
      nb0 = *(const uint4*)pb; nb1 = *(const uint4*)(pb + 8);
    }

    Frag16 a[2], b[4];
#pragma unroll
    for (int mi = 0; mi < 2; ++mi) {
      int row = wm * 32 + mi * 16 + l15;
      const unsigned int* p = (const unsigned int*)&As[cur][row * LPAD];
#pragma unroll
      for (int v = 0; v < 8; ++v) {
        int kk = ((v < 4) ? 2 * v : 2 * v + 8) + half * 8;  // ISA 16-bit A layout
        a[mi].u[v] = p[kk >> 1];
      }
    }
#pragma unroll
    for (int ni = 0; ni < 4; ++ni) {
      int col = wn * 64 + ni * 16 + l15;
      const unsigned int* p = (const unsigned int*)&Bs[cur][col * LPAD];
#pragma unroll
      for (int v = 0; v < 8; ++v) {
        int kk = ((v < 4) ? 2 * v : 2 * v + 8) + half * 8;  // ISA 16-bit B layout
        b[ni].u[v] = p[kk >> 1];
      }
    }

    if (pre) {  // stage next tile behind the WMMAs
      *(uint4*)&As[nxt][ldso] = na0; *(uint4*)&As[nxt][ldso + 8] = na1;
      *(uint4*)&Bs[nxt][ldso] = nb0; *(uint4*)&Bs[nxt][ldso + 8] = nb1;
    }

#pragma unroll
    for (int mi = 0; mi < 2; ++mi)
#pragma unroll
      for (int ni = 0; ni < 4; ++ni)
        acc[mi][ni] = __builtin_amdgcn_wmma_f32_16x16x32_bf16(
            false, a[mi].v, false, b[ni].v, (short)0, acc[mi][ni], false, false);

    __syncthreads();
  }

#pragma unroll
  for (int mi = 0; mi < 2; ++mi)
#pragma unroll
    for (int ni = 0; ni < 4; ++ni) {
      int col = bn + wn * 64 + ni * 16 + l15;
#pragma unroll
      for (int r = 0; r < 8; ++r) {
        int row = bm + wm * 32 + mi * 16 + r + half * 8;
        float v = acc[mi][ni][r];
        if (MODE == 1) {
          v += bias[col];
          v = 0.5f * v * (1.0f + erff(v * 0.70710678118654752f));  // exact GELU
        }
        if (MODE == 3) v += bias[col];
        size_t idx = (size_t)row * N + col;
        if (MODE == 0 || MODE == 1) Cbf[idx] = f2bf(v);
        else Xres[idx] = Xres[idx] + v;
      }
    }
}

// ---------------------------------------------------------------------------
// Windowed complex attention. One block per (b, window, head), 128 threads,
// thread i owns query row i. Scores Re(q·conj(k))*scale + rel_bias, softmax
// (2-pass: max, then exp-sum fused with P@V accumulation), normalize at end.
// qkv bf16 [MROWS,1536] cols: qr qi kr ki vr vi. obuf bf16 [MROWS,512].
// ---------------------------------------------------------------------------
__global__ __launch_bounds__(128) void attn_kernel(
    const unsigned short* __restrict__ qkv, const float* __restrict__ rel_bias,
    unsigned short* __restrict__ obuf) {
  __shared__ unsigned short KV[4][128 * 32];  // kr, ki, vr, vi
  __shared__ float bias_s[257];
  int tid = threadIdx.x;
  int bid = blockIdx.x;
  int h = bid & 7;
  int w = (bid >> 3) & 63;
  int b = bid >> 9;
  size_t rowbase = (size_t)b * SEQ + (size_t)w * WSZ;

  for (int i = tid; i < 257; i += 128) bias_s[i] = rel_bias[i];

  const unsigned short* rq = qkv + (rowbase + tid) * 1536;
  {
    const uint4* kr = (const uint4*)(rq + 512 + h * 32);
    const uint4* ki = (const uint4*)(rq + 768 + h * 32);
    const uint4* vr = (const uint4*)(rq + 1024 + h * 32);
    const uint4* vi = (const uint4*)(rq + 1280 + h * 32);
    uint4* d0 = (uint4*)&KV[0][tid * 32]; uint4* d1 = (uint4*)&KV[1][tid * 32];
    uint4* d2 = (uint4*)&KV[2][tid * 32]; uint4* d3 = (uint4*)&KV[3][tid * 32];
#pragma unroll
    for (int p = 0; p < 4; ++p) { d0[p] = kr[p]; d1[p] = ki[p]; d2[p] = vr[p]; d3[p] = vi[p]; }
  }
  float qr[32], qi[32];
#pragma unroll
  for (int d = 0; d < 32; ++d) {
    qr[d] = bf2f(rq[h * 32 + d]);
    qi[d] = bf2f(rq[256 + h * 32 + d]);
  }
  __syncthreads();

  // pass 1: row max
  float m = -1e30f;
  for (int j = 0; j < 128; ++j) {
    const unsigned short* kr = &KV[0][j * 32];
    const unsigned short* ki = &KV[1][j * 32];
    float s = 0.f;
#pragma unroll
    for (int d = 0; d < 32; ++d) s += qr[d] * bf2f(kr[d]) + qi[d] * bf2f(ki[d]);
    s = s * ATT_SCALE + bias_s[j - tid + 128];
    m = fmaxf(m, s);
  }
  // pass 2: exp-sum + unnormalized P@V
  float l = 0.f;
  float o_r[32], o_i[32];
#pragma unroll
  for (int d = 0; d < 32; ++d) { o_r[d] = 0.f; o_i[d] = 0.f; }
  for (int j = 0; j < 128; ++j) {
    const unsigned short* kr = &KV[0][j * 32];
    const unsigned short* ki = &KV[1][j * 32];
    float s = 0.f;
#pragma unroll
    for (int d = 0; d < 32; ++d) s += qr[d] * bf2f(kr[d]) + qi[d] * bf2f(ki[d]);
    s = s * ATT_SCALE + bias_s[j - tid + 128];
    float p = __expf(s - m);
    l += p;
    const unsigned short* vr = &KV[2][j * 32];
    const unsigned short* vi = &KV[3][j * 32];
#pragma unroll
    for (int d = 0; d < 32; ++d) {
      o_r[d] += p * bf2f(vr[d]);
      o_i[d] += p * bf2f(vi[d]);
    }
  }
  float inv = 1.0f / l;
  unsigned short* orow = obuf + (rowbase + tid) * 512;
#pragma unroll
  for (int d = 0; d < 32; ++d) {
    orow[h * 32 + d]       = f2bf(o_r[d] * inv);
    orow[256 + h * 32 + d] = f2bf(o_i[d] * inv);
  }
}

// ---------------------------------------------------------------------------
// Host-side orchestration
// ---------------------------------------------------------------------------
extern "C" void kernel_launch(void* const* d_in, const int* in_sizes, int n_in,
                              void* d_out, int out_size, void* d_ws, size_t ws_size,
                              hipStream_t stream) {
  const float* x_real = (const float*)d_in[0];
  const float* x_imag = (const float*)d_in[1];
  const float* Wq_r = (const float*)d_in[2];  const float* Wq_i = (const float*)d_in[3];
  const float* Wk_r = (const float*)d_in[4];  const float* Wk_i = (const float*)d_in[5];
  const float* Wv_r = (const float*)d_in[6];  const float* Wv_i = (const float*)d_in[7];
  const float* Wo_r = (const float*)d_in[8];  const float* Wo_i = (const float*)d_in[9];
  const float* rel_bias = (const float*)d_in[10];
  const float* norm1 = (const float*)d_in[11];
  const float* norm2 = (const float*)d_in[12];
  const float* W1_r = (const float*)d_in[13]; const float* b1_r = (const float*)d_in[14];
  const float* W1_i = (const float*)d_in[15]; const float* b1_i = (const float*)d_in[16];
  const float* W2_r = (const float*)d_in[17]; const float* b2_r = (const float*)d_in[18];
  const float* W2_i = (const float*)d_in[19]; const float* b2_i = (const float*)d_in[20];
  (void)in_sizes; (void)n_in; (void)out_size; (void)ws_size;

  char* ws = (char*)d_ws;
  size_t off = 0;
  float* X = (float*)(ws + off);               off += (size_t)MROWS * 512 * 4;   // 32 MB
  unsigned short* Abuf = (unsigned short*)(ws + off); off += (size_t)MROWS * 512 * 2; // 16 MB
  // QKV(48MB)+Obuf(16MB) live during attention; Hbuf(64MB) reuses the union.
  unsigned short* QKV  = (unsigned short*)(ws + off);
  unsigned short* Hbuf = (unsigned short*)(ws + off);
  unsigned short* Obuf = (unsigned short*)(ws + off + (size_t)MROWS * 1536 * 2);
  off += (size_t)MROWS * 2048 * 2;                                               // 64 MB
  unsigned short* WqkvP = (unsigned short*)(ws + off); off += (size_t)LAYERS * 512 * 1536 * 2;
  unsigned short* WoP   = (unsigned short*)(ws + off); off += (size_t)LAYERS * 512 * 512 * 2;
  unsigned short* W1P   = (unsigned short*)(ws + off); off += (size_t)LAYERS * 512 * 2048 * 2;
  unsigned short* W2P   = (unsigned short*)(ws + off); off += (size_t)LAYERS * 2048 * 512 * 2;
  float* B1P = (float*)(ws + off); off += (size_t)LAYERS * 2048 * 4;
  float* B2P = (float*)(ws + off); off += (size_t)LAYERS * 512 * 4;

  // 1) pack input activations
  pack_x_kernel<<<(MROWS * DMODEL + 255) / 256, 256, 0, stream>>>(x_real, x_imag, X);

  // 2) pack all weights/biases to bf16 complex-GEMM form (B stored as [N,K])
  for (int l = 0; l < LAYERS; ++l) {
    const size_t wDD = (size_t)DMODEL * DMODEL;       // 65536
    const size_t wDF = (size_t)DMODEL * FFDIM;        // 262144
    unsigned short* wq = WqkvP + (size_t)l * 512 * 1536;
    pack2_kernel<<<(4 * 256 * 256 + 255) / 256, 256, 0, stream>>>(Wq_r + l * wDD, Wq_i + l * wDD, wq, 256, 256, 512, 0);
    pack2_kernel<<<(4 * 256 * 256 + 255) / 256, 256, 0, stream>>>(Wk_r + l * wDD, Wk_i + l * wDD, wq, 256, 256, 512, 512);
    pack2_kernel<<<(4 * 256 * 256 + 255) / 256, 256, 0, stream>>>(Wv_r + l * wDD, Wv_i + l * wDD, wq, 256, 256, 512, 1024);
    pack2_kernel<<<(4 * 256 * 256 + 255) / 256, 256, 0, stream>>>(Wo_r + l * wDD, Wo_i + l * wDD, WoP + (size_t)l * 512 * 512, 256, 256, 512, 0);
    pack2_kernel<<<(4 * 256 * 1024 + 255) / 256, 256, 0, stream>>>(W1_r + l * wDF, W1_i + l * wDF, W1P + (size_t)l * 512 * 2048, 256, 1024, 512, 0);
    pack2_kernel<<<(4 * 1024 * 256 + 255) / 256, 256, 0, stream>>>(W2_r + l * wDF, W2_i + l * wDF, W2P + (size_t)l * 2048 * 512, 1024, 256, 2048, 0);
    packbias_kernel<<<(2048 + 255) / 256, 256, 0, stream>>>(b1_r + l * FFDIM, b1_i + l * FFDIM, B1P + (size_t)l * 2048, 1024);
    packbias_kernel<<<(512 + 255) / 256, 256, 0, stream>>>(b2_r + l * DMODEL, b2_i + l * DMODEL, B2P + (size_t)l * 512, 256);
  }

  // 3) layers
  for (int l = 0; l < LAYERS; ++l) {
    rmsnorm_kernel<<<MROWS, 128, 0, stream>>>(X, norm1 + l * DMODEL, Abuf);
    gemm_kernel<0><<<dim3(1536 / 128, MROWS / 128), 256, 0, stream>>>(
        Abuf, WqkvP + (size_t)l * 512 * 1536, QKV, nullptr, nullptr, 1536, 512);
    attn_kernel<<<NBATCH * (SEQ / WSZ) * NHEAD, 128, 0, stream>>>(
        QKV, rel_bias + l * (2 * WSZ + 1), Obuf);
    gemm_kernel<2><<<dim3(512 / 128, MROWS / 128), 256, 0, stream>>>(
        Obuf, WoP + (size_t)l * 512 * 512, nullptr, X, nullptr, 512, 512);
    rmsnorm_kernel<<<MROWS, 128, 0, stream>>>(X, norm2 + l * DMODEL, Abuf);
    gemm_kernel<1><<<dim3(2048 / 128, MROWS / 128), 256, 0, stream>>>(
        Abuf, W1P + (size_t)l * 512 * 2048, Hbuf, nullptr, B1P + (size_t)l * 2048, 2048, 512);
    gemm_kernel<3><<<dim3(512 / 128, MROWS / 128), 256, 0, stream>>>(
        Hbuf, W2P + (size_t)l * 2048 * 512, nullptr, X, B2P + (size_t)l * 512, 512, 2048);
  }

  // 4) unpack to [2, B, N, D] fp32
  unpack_kernel<<<(MROWS * DMODEL + 255) / 256, 256, 0, stream>>>(X, (float*)d_out);
}
